// Head_37263136260656
// MI455X (gfx1250) — matrix operand order repaired and checked by reading, not compiled, not run on hardware
//
#include <hip/hip_runtime.h>
#include <hip/hip_bf16.h>

// ---------------------------------------------------------------------------
// Fused single-head attention for MI455X (gfx1250, wave32, WMMA f16 16x16x32)
//   B=4, T=4 (BT=16 independent problems), S=2048, E=256, H=64
// Kernel 1: q/k/v projection (f32 in -> f16 out, f32 WMMA accumulate),
//           v stored transposed [H][S] for contiguous B-fragments later.
// Kernel 2: flash attention (online softmax) + fused output projection.
// ---------------------------------------------------------------------------

typedef __attribute__((ext_vector_type(16))) _Float16 v16h;
typedef __attribute__((ext_vector_type(8)))  _Float16 v8h;
typedef __attribute__((ext_vector_type(4)))  _Float16 v4h;
typedef __attribute__((ext_vector_type(8)))  float    v8f;

constexpr int BT = 16;   // B*T
constexpr int S  = 2048; // sequence length
constexpr int E  = 256;  // embedding dim
constexpr int H  = 64;   // head size

// Build a 16x32 f16 A-fragment from two 8-half contiguous groups.
// Layout used: lanes 0-15 hold row m=lane, k = {0..7} (lo) and {8..15} (hi);
//             lanes 16-31 hold k = {16..23} and {24..31}.  Caller passes the
//             per-lane base pointers (already offset by (lane>>4)*16).
__device__ inline v16h make_a_frag(const _Float16* lo, const _Float16* hi) {
  v8h l = *(const v8h*)lo;
  v8h h = *(const v8h*)hi;
  v16h r;
#pragma unroll
  for (int i = 0; i < 8; ++i) { r[i] = l[i]; r[8 + i] = h[i]; }
  return r;
}

// Butterfly reductions across the 16 lanes holding one C-matrix row
// (xor masks 1,2,4,8 never cross the 16-lane half boundary on wave32).
__device__ inline float halfmax16(float x) {
  x = fmaxf(x, __shfl_xor(x, 1, 32));
  x = fmaxf(x, __shfl_xor(x, 2, 32));
  x = fmaxf(x, __shfl_xor(x, 4, 32));
  x = fmaxf(x, __shfl_xor(x, 8, 32));
  return x;
}
__device__ inline float halfsum16(float x) {
  x += __shfl_xor(x, 1, 32);
  x += __shfl_xor(x, 2, 32);
  x += __shfl_xor(x, 4, 32);
  x += __shfl_xor(x, 8, 32);
  return x;
}

__device__ inline v8f wmma_f16(v16h a, v16h b, v8f c) {
  return __builtin_amdgcn_wmma_f32_16x16x32_f16(false, a, false, b,
                                                (short)0, c, false, false);
}

// ---------------------------------------------------------------------------
// Kernel 1: q/k/v projection.  Grid: (S/64, BT), 128 threads (4 waves).
// Each wave computes a [16 x 64] strip of q, k, v for its (bt, row-tile).
// ---------------------------------------------------------------------------
__global__ __launch_bounds__(128) void qkv_proj_kernel(
    const float* __restrict__ x,
    const float* __restrict__ Wq, const float* __restrict__ bq,
    const float* __restrict__ Wk, const float* __restrict__ bk,
    const float* __restrict__ Wv, const float* __restrict__ bv,
    _Float16* __restrict__ qout, _Float16* __restrict__ kout,
    _Float16* __restrict__ vtout)
{
  __shared__ __align__(32) _Float16 sW[3][H][32];  // per-chunk W slices, f16

  const int bt   = blockIdx.y;
  const int s0   = blockIdx.x * 64;
  const int tid  = threadIdx.x;
  const int wave = tid >> 5;
  const int lane = tid & 31;
  const int lh   = lane & 15;   // lane-in-half
  const int ls   = lane >> 4;   // half select

  const int    arow = s0 + wave * 16 + lh;                 // A-fragment row
  const float* xrow = x + ((size_t)bt * S + arow) * E;

  v8f acc[3][4];
#pragma unroll
  for (int m = 0; m < 3; ++m)
#pragma unroll
    for (int n = 0; n < 4; ++n) { v8f z = {}; acc[m][n] = z; }

  // Staging coordinates: each thread moves 4 consecutive f32 per pass,
  // 4 passes cover one 64x32 slice (2048 elems); fully unrolled, no selects.
  const int si0 = tid * 4;

  for (int ec = 0; ec < E; ec += 32) {
    __syncthreads();  // protect previous slice reads
#pragma unroll
    for (int p = 0; p < 4; ++p) {
      const int i = si0 + p * 512;
      const int r = i >> 5, c = i & 31;
      const size_t goff = (size_t)r * E + ec + c;
#pragma unroll
      for (int m = 0; m < 3; ++m) {
        const float* wsrc = (m == 0) ? Wq : (m == 1) ? Wk : Wv;
        const float4 w4 = *(const float4*)(wsrc + goff);
        v4h h4;
        h4[0] = (_Float16)w4.x; h4[1] = (_Float16)w4.y;
        h4[2] = (_Float16)w4.z; h4[3] = (_Float16)w4.w;
        *(v4h*)&sW[m][r][c] = h4;   // 8-byte ds_store
      }
    }
    __syncthreads();

    // A fragment of x (f32 -> f16), per-lane contiguous octets.
    v16h a;
#pragma unroll
    for (int j = 0; j < 8; ++j) {
      a[j]     = (_Float16)xrow[ec + ls * 16 + j];
      a[8 + j] = (_Float16)xrow[ec + ls * 16 + 8 + j];
    }

#pragma unroll
    for (int m = 0; m < 3; ++m) {
#pragma unroll
      for (int n = 0; n < 4; ++n) {
        // B fragment: col n*16+lh, 16 contiguous k starting at ls*16.
        v16h b = *(const v16h*)&sW[m][n * 16 + lh][ls * 16];
        acc[m][n] = wmma_f16(a, b, acc[m][n]);
      }
    }
  }

  // Bias + stores.  C layout: vgpr j -> row ls*8+j, col lh (per 16-tile n).
#pragma unroll
  for (int n = 0; n < 4; ++n) {
    const int   h   = n * 16 + lh;
    const float bq_ = bq[h], bk_ = bk[h], bv_ = bv[h];
#pragma unroll
    for (int j = 0; j < 8; ++j) {
      const int    srow = s0 + wave * 16 + ls * 8 + j;
      const size_t rh   = ((size_t)bt * S + srow) * H + h;
      qout[rh] = (_Float16)(acc[0][n][j] + bq_);
      kout[rh] = (_Float16)(acc[1][n][j] + bk_);
      vtout[((size_t)bt * H + h) * S + srow] = (_Float16)(acc[2][n][j] + bv_);
    }
  }
}

// ---------------------------------------------------------------------------
// Kernel 2: flash attention + output projection.  Grid: (S/64, BT), 128 thr.
// Each wave owns 16 query rows; loops keys in 32-wide steps.
// ---------------------------------------------------------------------------
__global__ __launch_bounds__(128) void attn_kernel(
    const _Float16* __restrict__ qin, const _Float16* __restrict__ kin,
    const _Float16* __restrict__ vtin, const unsigned char* __restrict__ mask,
    const float* __restrict__ Wo, const float* __restrict__ bo,
    float* __restrict__ out)
{
  __shared__ __align__(32) _Float16 sP[4][16][64];  // per-wave C->A scratch

  const int bt   = blockIdx.y;
  const int q0   = blockIdx.x * 64;
  const int tid  = threadIdx.x;
  const int wave = tid >> 5;
  const int lane = tid & 31;
  const int lh   = lane & 15;
  const int ls   = lane >> 4;

  const float scale = 0.0625f;  // 1/sqrt(E) = 1/16 (reference scales by E!)

  // Q fragments (A layout) for rows q0 + wave*16 + (lane&15).
  const _Float16* qrow = qin + ((size_t)bt * S + q0 + wave * 16 + lh) * H;
  const v16h aq0 = make_a_frag(qrow + ls * 16,      qrow + ls * 16 + 8);
  const v16h aq1 = make_a_frag(qrow + 32 + ls * 16, qrow + 40 + ls * 16);

  v8f O[4];
#pragma unroll
  for (int n = 0; n < 4; ++n) { v8f z = {}; O[n] = z; }
  float m_run[8], ell[8];
#pragma unroll
  for (int j = 0; j < 8; ++j) { m_run[j] = -INFINITY; ell[j] = 0.0f; }

  const unsigned char* mbase = mask + (size_t)bt * S * S;
  const _Float16*      kbt   = kin  + (size_t)bt * S * H;
  const _Float16*      vbt   = vtin + (size_t)bt * H * S;

  for (int kb = 0; kb < S; kb += 32) {
    if (kb + 32 < S) {  // emits global_prefetch_b8
      __builtin_prefetch(kbt + (size_t)(kb + 32 + lh) * H, 0, 1);
      __builtin_prefetch(vbt + (size_t)lh * S + kb + 32, 0, 1);
    }

    // Scores: two 16-key tiles, K-dim 64 = 2 WMMAs each.
    v8f s[2];
#pragma unroll
    for (int t = 0; t < 2; ++t) {
      const _Float16* krow = kbt + (size_t)(kb + 16 * t + lh) * H;
      v16h b0 = *(const v16h*)(krow + ls * 16);        // h 0..31 chunk
      v16h b1 = *(const v16h*)(krow + 32 + ls * 16);   // h 32..63 chunk
      v8f c = {};
      c = wmma_f16(aq0, b0, c);
      c = wmma_f16(aq1, b1, c);
      s[t] = c;
    }

    // Scale + mask (True = masked out -> -1e7, exactly like reference).
#pragma unroll
    for (int j = 0; j < 8; ++j) {
      const unsigned char* mrow =
          mbase + (size_t)(q0 + wave * 16 + ls * 8 + j) * S + kb;
      float s0v = s[0][j] * scale;
      float s1v = s[1][j] * scale;
      if (mrow[lh])      s0v = -10000000.0f;
      if (mrow[16 + lh]) s1v = -10000000.0f;
      s[0][j] = s0v; s[1][j] = s1v;
    }

    // Online softmax update.
    float corr[8];
#pragma unroll
    for (int j = 0; j < 8; ++j) {
      float rm = halfmax16(fmaxf(s[0][j], s[1][j]));
      float mn = fmaxf(m_run[j], rm);
      corr[j]  = __expf(m_run[j] - mn);   // exp(-inf)=0 on first tile
      m_run[j] = mn;
      float p0 = __expf(s[0][j] - mn);
      float p1 = __expf(s[1][j] - mn);
      s[0][j] = p0; s[1][j] = p1;
      ell[j] = ell[j] * corr[j] + halfsum16(p0 + p1);
    }
#pragma unroll
    for (int n = 0; n < 4; ++n)
#pragma unroll
      for (int j = 0; j < 8; ++j) O[n][j] *= corr[j];

    // Re-lay P from C layout to A layout through this wave's LDS strip
    // (same-wave DS ops are in-order; no barrier needed).
#pragma unroll
    for (int j = 0; j < 8; ++j) {
      sP[wave][ls * 8 + j][lh]      = (_Float16)s[0][j];
      sP[wave][ls * 8 + j][16 + lh] = (_Float16)s[1][j];
    }
    const _Float16* prow = &sP[wave][lh][0];
    const v16h ap = make_a_frag(prow + ls * 16, prow + ls * 16 + 8);

    // P @ V: v stored transposed -> 16 contiguous f16 per lane per fragment.
#pragma unroll
    for (int n = 0; n < 4; ++n) {
      const _Float16* vrow = vbt + (size_t)(n * 16 + lh) * S + kb + ls * 16;
      v16h bv_ = *(const v16h*)vrow;
      O[n] = wmma_f16(ap, bv_, O[n]);
    }
  }

  // Normalize.
#pragma unroll
  for (int j = 0; j < 8; ++j) {
    const float inv = 1.0f / ell[j];
#pragma unroll
    for (int n = 0; n < 4; ++n) O[n][j] *= inv;
  }

  // Fused output projection: O (C layout) -> LDS -> A fragments, x Wo^T.
#pragma unroll
  for (int n = 0; n < 4; ++n)
#pragma unroll
    for (int j = 0; j < 8; ++j)
      sP[wave][ls * 8 + j][n * 16 + lh] = (_Float16)O[n][j];
  const _Float16* orow = &sP[wave][lh][0];
  const v16h ao0 = make_a_frag(orow + ls * 16,      orow + ls * 16 + 8);
  const v16h ao1 = make_a_frag(orow + 32 + ls * 16, orow + 40 + ls * 16);

#pragma unroll
  for (int n = 0; n < 4; ++n) {
    // B[k=h][g]: 16 contiguous f32 of Wo row g, converted to f16.
    const float* w0 = Wo + (size_t)(n * 16 + lh) * H + ls * 16;
    v16h bw0, bw1;
#pragma unroll
    for (int i = 0; i < 16; ++i) {
      bw0[i] = (_Float16)w0[i];        // h chunk 0..31
      bw1[i] = (_Float16)w0[32 + i];   // h chunk 32..63
    }
    v8f oc = {};
    oc = wmma_f16(ao0, bw0, oc);
    oc = wmma_f16(ao1, bw1, oc);
    const float bo_ = bo[n * 16 + lh];
#pragma unroll
    for (int j = 0; j < 8; ++j) {
      out[((size_t)bt * S + q0 + wave * 16 + ls * 8 + j) * H + n * 16 + lh] =
          oc[j] + bo_;
    }
  }
}

// ---------------------------------------------------------------------------
extern "C" void kernel_launch(void* const* d_in, const int* in_sizes, int n_in,
                              void* d_out, int out_size, void* d_ws, size_t ws_size,
                              hipStream_t stream) {
  const float*         x    = (const float*)d_in[0];
  const unsigned char* mask = (const unsigned char*)d_in[1];  // jnp bool = 1B
  const float*         Wq   = (const float*)d_in[2];
  const float*         bq   = (const float*)d_in[3];
  const float*         Wk   = (const float*)d_in[4];
  const float*         bk   = (const float*)d_in[5];
  const float*         Wv   = (const float*)d_in[6];
  const float*         bv   = (const float*)d_in[7];
  const float*         Wo   = (const float*)d_in[8];
  const float*         bo   = (const float*)d_in[9];
  float*               out  = (float*)d_out;

  // Workspace: q, k (row-major f16) and v^T (f16), 4 MB each = 12 MB total.
  _Float16* qws  = (_Float16*)d_ws;
  _Float16* kws  = qws + (size_t)BT * S * H;
  _Float16* vtws = kws + (size_t)BT * S * H;

  dim3 grid(S / 64, BT);
  qkv_proj_kernel<<<grid, 128, 0, stream>>>(x, Wq, bq, Wk, bk, Wv, bv,
                                            qws, kws, vtws);
  attn_kernel<<<grid, 128, 0, stream>>>(qws, kws, vtws, mask, Wo, bo, out);
}
